// GATClassifier_34815004901853
// MI455X (gfx1250) — matrix operand order repaired and checked by reading, not compiled, not run on hardware
//
#include <hip/hip_runtime.h>
#include <hip/hip_bf16.h>
#include <math.h>

// ---------------- problem constants (match reference) ----------------
#define NN 50000      // nodes
#define EE 800000     // edges (before self loops)
#define ET (EE + NN)  // edges incl self loops
#define GG 512        // graphs
#define HH 4          // heads (layers 0,1)
#define CC 32         // channels per head
#define HC (HH * CC)  // 128
#define NEG_SLOPE 0.2f
#define BN_EPS 1e-5f

typedef __attribute__((ext_vector_type(16))) _Float16 v16h;
typedef __attribute__((ext_vector_type(8)))  _Float16 v8h;
typedef __attribute__((ext_vector_type(8)))  float    v8f;

// monotone uint key encoding for float atomic-max
__device__ __forceinline__ unsigned fkey(float f) {
    unsigned b = __float_as_uint(f);
    return (b & 0x80000000u) ? ~b : (b | 0x80000000u);
}
__device__ __forceinline__ float fdec(unsigned k) {
    return (k & 0x80000000u) ? __uint_as_float(k & 0x7fffffffu) : __uint_as_float(~k);
}

// ---------------- utility: zero fill (graph-capture safe init) ----------------
__global__ void k_zero(unsigned* __restrict__ p, long long n) {
    long long i = (long long)blockIdx.x * blockDim.x + threadIdx.x;
    if (i < n) p[i] = 0u;
}

// ---------------- f32 -> f16 staging with K padding ----------------
// dst[row*Kpad + k] = (k < din) ? f16(src[row*din + k]) : 0
__global__ void k_cvt_feat(const float* __restrict__ src, _Float16* __restrict__ dst,
                           int din, int Kpad, long long total) {
    long long i = (long long)blockIdx.x * blockDim.x + threadIdx.x;
    if (i >= total) return;
    int k = (int)(i % Kpad);
    long long row = i / Kpad;
    dst[i] = (k < din) ? (_Float16)src[row * din + k] : (_Float16)0.f;
}

// W [din, dout] row-major f32  ->  wh [dout, Kpad] f16 (transposed, K padded)
__global__ void k_cvt_w(const float* __restrict__ W, _Float16* __restrict__ wh,
                        int din, int dout, int Kpad) {
    int i = blockIdx.x * blockDim.x + threadIdx.x;
    if (i >= dout * Kpad) return;
    int k = i % Kpad;
    int n = i / Kpad;
    wh[i] = (k < din) ? (_Float16)W[k * dout + n] : (_Float16)0.f;
}

// ---------------- WMMA GEMM: D[N, dout] = A[N, Kpad] * B^T (wh is [dout,Kpad]) ----------------
// One wave computes one 16x16 output tile via v_wmma_f32_16x16x32_f16.
__global__ void k_gemm_wmma(const _Float16* __restrict__ Ah,
                            const _Float16* __restrict__ Bh,
                            float* __restrict__ D,
                            int Kpad, int dout) {
    const int wave = (blockIdx.x * blockDim.x + threadIdx.x) >> 5;
    const int lane = threadIdx.x & 31;
    const int ntiles = dout >> 4;
    const int mtiles = NN >> 4;          // 3125, exact
    if (wave >= mtiles * ntiles) return; // wave-uniform exit; EXEC all-1s for WMMA
    const int mt = wave / ntiles;
    const int nt = wave - mt * ntiles;
    const int half = lane >> 4;          // 0: lanes 0-15, 1: lanes 16-31
    const int mrow = (mt << 4) + (lane & 15);
    const int ncol = (nt << 4) + (lane & 15);

    const _Float16* arow = Ah + (long long)mrow * Kpad;
    const _Float16* bcol = Bh + (long long)ncol * Kpad;

    v8f acc = {};
    for (int kk = 0; kk < Kpad; kk += 32) {
        // A 16x32 f16 layout: lanes<16 hold K {kk..kk+7, kk+16..kk+23}; lanes>=16 hold {+8..+15, +24..+31}
        v8h a0 = *(const v8h*)(arow + kk + (half << 3));
        v8h a1 = *(const v8h*)(arow + kk + 16 + (half << 3));
        // B 32x16 f16 layout: lanes<16 hold K {kk..kk+7, kk+16..kk+23} of column ncol; lanes>=16 hold {+8..+15, +24..+31}
        v8h b0 = *(const v8h*)(bcol + kk + (half << 3));
        v8h b1 = *(const v8h*)(bcol + kk + 16 + (half << 3));
        v16h a, b;
#pragma unroll
        for (int i = 0; i < 8; ++i) {
            a[i] = a0[i]; a[8 + i] = a1[i];
            b[i] = b0[i]; b[8 + i] = b1[i];
        }
        acc = __builtin_amdgcn_wmma_f32_16x16x32_f16(false, a, false, b,
                                                     (short)0, acc, false, false);
    }
    // C/D layout: lanes<16 rows 0..7 in vgpr 0..7 (col=lane); lanes>=16 rows 8..15
    float* dp = D + ((long long)(mt << 4) + (half << 3)) * dout + ncol;
#pragma unroll
    for (int i = 0; i < 8; ++i) dp[(long long)i * dout] = acc[i];
}

// ---------------- attention coefficients: al_s/al_d [N, heads] ----------------
__global__ void k_attn_coef(const float* __restrict__ h,
                            const float* __restrict__ a_s, const float* __restrict__ a_d,
                            float* __restrict__ al_s, float* __restrict__ al_d, int heads) {
    int t = blockIdx.x * blockDim.x + threadIdx.x;
    if (t >= NN * heads) return;
    int n = t / heads, hh = t - n * heads;
    const float* hp = h + (long long)n * heads * CC + hh * CC;
    const float* as = a_s + hh * CC;
    const float* ad = a_d + hh * CC;
    float s = 0.f, d = 0.f;
#pragma unroll
    for (int c = 0; c < CC; ++c) { float v = hp[c]; s += v * as[c]; d += v * ad[c]; }
    al_s[t] = s; al_d[t] = d;
}

__device__ __forceinline__ void edge_endpoints(const int* ei, int e, int& src, int& dst) {
    if (e < EE) { src = ei[e]; dst = ei[EE + e]; }
    else        { src = e - EE; dst = e - EE; }   // self loops
}

// ---------------- per-dst segment max (softmax stabilization) ----------------
__global__ void k_edge_max(const int* __restrict__ ei,
                           const float* __restrict__ al_s, const float* __restrict__ al_d,
                           unsigned* __restrict__ mkey, int heads) {
    int e = blockIdx.x * blockDim.x + threadIdx.x;
    if (e >= ET) return;
    int src, dst; edge_endpoints(ei, e, src, dst);
    for (int hh = 0; hh < heads; ++hh) {
        float t = al_s[src * heads + hh] + al_d[dst * heads + hh];
        t = t > 0.f ? t : NEG_SLOPE * t;
        atomicMax(&mkey[dst * heads + hh], fkey(t));
    }
}

// ---------------- per-dst exp-sum ----------------
__global__ void k_edge_sum(const int* __restrict__ ei,
                           const float* __restrict__ al_s, const float* __restrict__ al_d,
                           const unsigned* __restrict__ mkey, float* __restrict__ ssum, int heads) {
    int e = blockIdx.x * blockDim.x + threadIdx.x;
    if (e >= ET) return;
    int src, dst; edge_endpoints(ei, e, src, dst);
    for (int hh = 0; hh < heads; ++hh) {
        float t = al_s[src * heads + hh] + al_d[dst * heads + hh];
        t = t > 0.f ? t : NEG_SLOPE * t;
        float p = __expf(t - fdec(mkey[dst * heads + hh]));
        atomicAdd(&ssum[dst * heads + hh], p);
    }
}

// ---------------- weighted aggregation: one wave per edge, lane = channel ----------------
__global__ void k_edge_aggr(const int* __restrict__ ei,
                            const float* __restrict__ al_s, const float* __restrict__ al_d,
                            const unsigned* __restrict__ mkey, const float* __restrict__ ssum,
                            const float* __restrict__ h, float* __restrict__ agg, int heads) {
    long long gid = (long long)blockIdx.x * blockDim.x + threadIdx.x;
    int e = (int)(gid >> 5);
    int lane = (int)(gid & 31);
    if (e >= ET) return;
    int src, dst; edge_endpoints(ei, e, src, dst);
    const float* hs = h + (long long)src * heads * CC;
    float* ad = agg + (long long)dst * heads * CC;
    __builtin_prefetch(hs, 0, 0);                 // global_prefetch_b8: pull src row early
    for (int hh = 0; hh < heads; ++hh) {
        float t = al_s[src * heads + hh] + al_d[dst * heads + hh];
        t = t > 0.f ? t : NEG_SLOPE * t;
        float alpha = __expf(t - fdec(mkey[dst * heads + hh])) /
                      (ssum[dst * heads + hh] + 1e-16f);
        atomicAdd(&ad[hh * CC + lane], hs[hh * CC + lane] * alpha);
    }
}

// ---------------- bias + eval BatchNorm + ELU (in place) ----------------
__global__ void k_post(float* __restrict__ agg,
                       const float* __restrict__ bias,
                       const float* __restrict__ bng, const float* __restrict__ bnb,
                       const float* __restrict__ bnm, const float* __restrict__ bnv, int dim) {
    long long i = (long long)blockIdx.x * blockDim.x + threadIdx.x;
    if (i >= (long long)NN * dim) return;
    int d = (int)(i % dim);
    float x = agg[i] + bias[d];
    x = (x - bnm[d]) * rsqrtf(bnv[d] + BN_EPS) * bng[d] + bnb[d];
    agg[i] = x > 0.f ? x : (__expf(x) - 1.f);
}

// ---------------- graph pooling: mean (sum+cnt) and max ----------------
__global__ void k_pool(const float* __restrict__ feat, const int* __restrict__ batch,
                       float* __restrict__ gsum, unsigned* __restrict__ gmax,
                       float* __restrict__ gcnt) {
    long long i = (long long)blockIdx.x * blockDim.x + threadIdx.x;
    if (i >= (long long)NN * CC) return;
    int n = (int)(i >> 5), c = (int)(i & 31);
    int g = batch[n];
    float v = feat[(long long)n * CC + c];
    atomicAdd(&gsum[g * CC + c], v);
    atomicMax(&gmax[g * CC + c], fkey(v));
    if (c == 0) atomicAdd(&gcnt[g], 1.0f);
}

// ---------------- readout MLP: [mean|max|u] @ l1w, relu, @ l2w, sigmoid ----------------
__global__ void k_mlp(const float* __restrict__ gsum, const unsigned* __restrict__ gmax,
                      const float* __restrict__ gcnt, const float* __restrict__ u,
                      const float* __restrict__ l1w, const float* __restrict__ l1b,
                      const float* __restrict__ l2w, const float* __restrict__ l2b,
                      float* __restrict__ out) {
    int g = blockIdx.x * blockDim.x + threadIdx.x;
    if (g >= GG) return;
    float comb[2 * CC + 10];
    float inv = 1.f / gcnt[g];
#pragma unroll
    for (int c = 0; c < CC; ++c) comb[c] = gsum[g * CC + c] * inv;
#pragma unroll
    for (int c = 0; c < CC; ++c) comb[CC + c] = fdec(gmax[g * CC + c]);
#pragma unroll
    for (int i = 0; i < 10; ++i) comb[2 * CC + i] = u[g * 10 + i];
    float acc = l2b[0];
    for (int j = 0; j < CC; ++j) {
        float hsum = l1b[j];
#pragma unroll 2
        for (int i = 0; i < 2 * CC + 10; ++i) hsum += comb[i] * l1w[i * CC + j];
        hsum = hsum > 0.f ? hsum : 0.f;
        acc += hsum * l2w[j];
    }
    out[g] = 1.f / (1.f + __expf(-acc));
}

// ---------------- host orchestration ----------------
static inline long long cdivll(long long a, long long b) { return (a + b - 1) / b; }

extern "C" void kernel_launch(void* const* d_in, const int* in_sizes, int n_in,
                              void* d_out, int out_size, void* d_ws, size_t ws_size,
                              hipStream_t stream) {
    (void)in_sizes; (void)n_in; (void)out_size; (void)ws_size;
    const float* x     = (const float*)d_in[0];
    const int*   ei    = (const int*)d_in[1];
    const int*   batch = (const int*)d_in[2];
    const float* u     = (const float*)d_in[3];
    const float* l1w = (const float*)d_in[28];
    const float* l1b = (const float*)d_in[29];
    const float* l2w = (const float*)d_in[30];
    const float* l2b = (const float*)d_in[31];
    float* out = (float*)d_out;

    // workspace carve-out (256B aligned); total ~67 MB
    char* ws = (char*)d_ws;
    size_t off = 0;
    auto carve = [&](size_t bytes) -> void* {
        void* p = ws + off;
        off += (bytes + 255) & ~(size_t)255;
        return p;
    };
    float*     h_buf = (float*)    carve((size_t)NN * HC * 4);  // GEMM output
    float*     agg   = (float*)    carve((size_t)NN * HC * 4);  // aggregated / layer output
    _Float16*  featH = (_Float16*) carve((size_t)NN * HC * 2);  // f16 staged input
    _Float16*  wH    = (_Float16*) carve((size_t)HC * HC * 2);  // f16 W^T
    float*     al_s  = (float*)    carve((size_t)NN * HH * 4);
    float*     al_d  = (float*)    carve((size_t)NN * HH * 4);
    unsigned*  mkey  = (unsigned*) carve((size_t)NN * HH * 4);
    float*     ssum  = (float*)    carve((size_t)NN * HH * 4);
    float*     gsum  = (float*)    carve((size_t)GG * CC * 4);
    unsigned*  gmax  = (unsigned*) carve((size_t)GG * CC * 4);
    float*     gcnt  = (float*)    carve((size_t)GG * 4);

    const int B = 256;
    struct Layer { int din, dout, heads, Kpad, base; };
    const Layer layers[3] = {
        {4,   HC, HH, 32,  4},
        {HC,  HC, HH, 128, 12},
        {HC,  CC, 1,  128, 20},
    };

    for (int l = 0; l < 3; ++l) {
        const Layer& L = layers[l];
        const float* Wl  = (const float*)d_in[L.base + 0];
        const float* asl = (const float*)d_in[L.base + 1];
        const float* adl = (const float*)d_in[L.base + 2];
        const float* bl  = (const float*)d_in[L.base + 3];
        const float* bng = (const float*)d_in[L.base + 4];
        const float* bnb = (const float*)d_in[L.base + 5];
        const float* bnm = (const float*)d_in[L.base + 6];
        const float* bnv = (const float*)d_in[L.base + 7];
        const float* input = (l == 0) ? x : agg;

        // 1. stage input as f16 (K padded to multiple of 32)
        long long fh = (long long)NN * L.Kpad;
        k_cvt_feat<<<(unsigned)cdivll(fh, B), B, 0, stream>>>(input, featH, L.din, L.Kpad, fh);
        // 2. stage W^T as f16
        int wh = L.dout * L.Kpad;
        k_cvt_w<<<(unsigned)cdivll(wh, B), B, 0, stream>>>(Wl, wH, L.din, L.dout, L.Kpad);
        // 3. WMMA GEMM: h = input @ W
        long long waves = (long long)(NN / 16) * (L.dout / 16);
        k_gemm_wmma<<<(unsigned)cdivll(waves * 32, B), B, 0, stream>>>(featH, wH, h_buf, L.Kpad, L.dout);
        // 4. zero accumulators
        long long nh = (long long)NN * L.heads;
        k_zero<<<(unsigned)cdivll(nh, B), B, 0, stream>>>(mkey, nh);
        k_zero<<<(unsigned)cdivll(nh, B), B, 0, stream>>>((unsigned*)ssum, nh);
        long long nd = (long long)NN * L.dout;
        k_zero<<<(unsigned)cdivll(nd, B), B, 0, stream>>>((unsigned*)agg, nd);
        // 5. attention coefficients
        k_attn_coef<<<(unsigned)cdivll(nh, B), B, 0, stream>>>(h_buf, asl, adl, al_s, al_d, L.heads);
        // 6/7/8. per-dst softmax over edges + weighted scatter-add
        k_edge_max <<<(unsigned)cdivll(ET, B), B, 0, stream>>>(ei, al_s, al_d, mkey, L.heads);
        k_edge_sum <<<(unsigned)cdivll(ET, B), B, 0, stream>>>(ei, al_s, al_d, mkey, ssum, L.heads);
        k_edge_aggr<<<(unsigned)cdivll((long long)ET * 32, B), B, 0, stream>>>(
            ei, al_s, al_d, mkey, ssum, h_buf, agg, L.heads);
        // 9. bias + BN + ELU (in place -> becomes next layer input)
        k_post<<<(unsigned)cdivll(nd, B), B, 0, stream>>>(agg, bl, bng, bnb, bnm, bnv, L.dout);
    }

    // pooling + readout
    k_zero<<<(unsigned)cdivll((long long)GG * CC, B), B, 0, stream>>>((unsigned*)gsum, (long long)GG * CC);
    k_zero<<<(unsigned)cdivll((long long)GG * CC, B), B, 0, stream>>>(gmax, (long long)GG * CC);
    k_zero<<<(unsigned)cdivll(GG, B), B, 0, stream>>>((unsigned*)gcnt, GG);
    k_pool<<<(unsigned)cdivll((long long)NN * CC, B), B, 0, stream>>>(agg, batch, gsum, gmax, gcnt);
    k_mlp<<<(unsigned)cdivll(GG, B), B, 0, stream>>>(gsum, gmax, gcnt, u, l1w, l1b, l2w, l2b, out);
}
